// V18AlgebraMultistepModelA_54013508714940
// MI455X (gfx1250) — matrix operand order repaired and checked by reading, not compiled, not run on hardware
//
#include <hip/hip_runtime.h>
#include <hip/hip_bf16.h>
#include <math.h>

typedef __attribute__((ext_vector_type(16))) _Float16 v16h;
typedef __attribute__((ext_vector_type(8)))  float    v8f;

namespace {
constexpr int V_    = 32;
constexpr int N_    = 4096;
constexpr int Y_    = 16;
constexpr int HID_  = 128;
constexpr int OBS_  = 64;
constexpr int NW_   = N_ / 32;       // 128 bitmask words per batch
constexpr int SPARSE_MAX = 1024;     // above this, dense streaming wins
}

__device__ __forceinline__ float gelu_exact(float x) {
    return 0.5f * x * (1.0f + erff(x * 0.70710678118654752f));
}

// A fragment (16x32 f16) from an LDS f32 tile, per ISA 16-bit A layout:
// lane L: M = L&15; VGPR v<4 holds K = 2v,2v+1 (+8 if L>=16); v>=4 holds K=16+2(v-4).. (+8)
__device__ __forceinline__ v16h load_a_frag(const float* src, int ld, int kc, int lane) {
    int M  = lane & 15;
    int hs = (lane >> 4) * 8;
    v16h a;
#pragma unroll
    for (int v = 0; v < 8; ++v) {
        int kb = ((v < 4) ? (2 * v) : (16 + 2 * (v - 4))) + hs;
        a[2 * v]     = (_Float16)src[M * ld + kc + kb];
        a[2 * v + 1] = (_Float16)src[M * ld + kc + kb + 1];
    }
    return a;
}

// B fragment (32x16 f16) from a row-major KxNfull f32 weight matrix:
// lane L: VGPR v<4 holds K = kc+2v,2v+1 (+8 if L>=16); v>=4 holds K=kc+16+2(v-4).. (+8). N = n0 + (lane&15).
// Rows >= krows and cols >= ncols are zero-padded.
__device__ __forceinline__ v16h load_b_frag(const float* W, int nfull, int krows, int ncols,
                                            int kc, int n0, int lane) {
    int n  = n0 + (lane & 15);
    int kb = kc + ((lane >> 4) ? 8 : 0);
    bool ncl = (n < ncols);
    v16h bb;
#pragma unroll
    for (int v = 0; v < 8; ++v) {
        int k0 = kb + ((v < 4) ? (2 * v) : (16 + 2 * (v - 4)));
        float w0 = (ncl && k0     < krows) ? W[k0       * nfull + n] : 0.f;
        float w1 = (ncl && k0 + 1 < krows) ? W[(k0 + 1) * nfull + n] : 0.f;
        bb[2 * v]     = (_Float16)w0;
        bb[2 * v + 1] = (_Float16)w1;
    }
    return bb;
}

// ---------------------------------------------------------------------------
// Phase 1: per-batch candidate mask (reads only rows {0, a0..aT-1}), class
// scatter (p_sig), entropy/top2/mass, candidate bitmask + compact list + count.
// ---------------------------------------------------------------------------
__global__ void phase1_kernel(const int* __restrict__ tables, const int* __restrict__ sigma,
                              const int* __restrict__ base_obs, const int* __restrict__ actions,
                              const int* __restrict__ responses, int T,
                              unsigned int* __restrict__ candWs, float* __restrict__ psig,
                              float* __restrict__ scal, int* __restrict__ cntWs,
                              unsigned short* __restrict__ listWs, int buildList)
{
    __shared__ unsigned int mask[NW_];
    __shared__ int clsCnt[Y_];
    __shared__ int lcnt;

    int b   = blockIdx.x;
    int tid = threadIdx.x;

    for (int i = tid; i < NW_; i += blockDim.x) mask[i] = 0u;
    if (tid < Y_) clsCnt[tid] = 0;
    if (tid == 0) lcnt = 0;
    __syncthreads();

    int base = base_obs[b];
    int act[8], rsp[8];
    for (int i = 0; i < T && i < 8; ++i) {
        int a = actions[b * T + i];
        act[i] = a < 0 ? 0 : (a > V_ - 1 ? V_ - 1 : a);
        rsp[i] = responses[b * T + i];
    }

    const int* tabB = tables + (size_t)b * V_ * N_;
    for (int n = tid; n < N_; n += blockDim.x) {
        bool m = (tabB[n] == base);
        for (int i = 0; i < T && i < 8; ++i)
            m = m && (tabB[act[i] * N_ + n] == rsp[i]);
        if (m) {
            atomicOr(&mask[n >> 5], 1u << (n & 31));
            atomicAdd(&clsCnt[sigma[(size_t)b * N_ + n]], 1);
            int slot = atomicAdd(&lcnt, 1);
            if (buildList) listWs[(size_t)b * N_ + slot] = (unsigned short)n;
        }
    }
    __syncthreads();

    if (tid == 0) {
        int total = 0;
        for (int y = 0; y < Y_; ++y) total += clsCnt[y];
        float inv = 1.0f / (float)(total > 0 ? total : 1);
        float ent = 0.f, m1 = -1.f, m2 = -1.f;
        for (int y = 0; y < Y_; ++y) {
            float p = (float)clsCnt[y] * inv;
            psig[b * Y_ + y] = p;
            float pc = p > 1e-9f ? p : 1e-9f;
            ent -= pc * logf(pc);
            if (p > m1) { m2 = m1; m1 = p; }
            else if (p > m2) { m2 = p; }
        }
        scal[b * 4 + 0] = ent;
        scal[b * 4 + 1] = m1;
        scal[b * 4 + 2] = m2;
        scal[b * 4 + 3] = (total > 0) ? inv : 0.f;   // max(state_dist)
        cntWs[b] = total;
    }
    for (int i = tid; i < NW_; i += blockDim.x) candWs[b * NW_ + i] = mask[i];
}

// ---------------------------------------------------------------------------
// Phase 2: distinct-count per (b,v). One wave per pair.
//   cnt==0      -> uniq = 0, zero table traffic (the expected case: E[cnt]~0.016)
//   cnt small   -> sparse gather of only candidate columns
//   cnt large   -> dense b128 streaming of the whole row at HBM rate
// ---------------------------------------------------------------------------
__global__ void phase2_kernel(const int* __restrict__ tables, const unsigned int* __restrict__ candWs,
                              const int* __restrict__ cntWs, const unsigned short* __restrict__ listWs,
                              float* __restrict__ uniq, int haveList, int numPairs)
{
    int wavesPerBlock = blockDim.x >> 5;
    int wid  = blockIdx.x * wavesPerBlock + (threadIdx.x >> 5);
    if (wid >= numPairs) return;
    int lane = threadIdx.x & 31;
    int b    = wid >> 5;                       // V_ == 32, pair index == b*V+v
    const int* row = tables + (size_t)wid * N_;

    int cnt = cntWs[b];
    if (cnt == 0) {
        if (lane == 0) uniq[wid] = 0.f;
        return;
    }

    unsigned long long pres = 0ull;
    if (haveList && cnt <= SPARSE_MAX) {
        const unsigned short* lst = listWs + (size_t)b * N_;
        for (int i = lane; i < cnt; i += 32) {
            int n = lst[i];
            pres |= 1ull << (row[n] & 63);
        }
    } else {
        const int4* row4 = reinterpret_cast<const int4*>(row);
        const unsigned int* cm = candWs + b * NW_;
        for (int it = 0; it < N_ / 128; ++it) {
            int4 vl = row4[it * 32 + lane];                       // coalesced 512B/wave
            unsigned int w    = cm[it * 4 + (lane >> 3)];
            unsigned int bits = (w >> ((lane & 7) * 4)) & 0xFu;
            if (bits & 1u) pres |= 1ull << (vl.x & 63);
            if (bits & 2u) pres |= 1ull << (vl.y & 63);
            if (bits & 4u) pres |= 1ull << (vl.z & 63);
            if (bits & 8u) pres |= 1ull << (vl.w & 63);
        }
    }
    // wave32 OR-reduction
    for (int off = 16; off >= 1; off >>= 1) {
        unsigned int lo = (unsigned int)pres;
        unsigned int hi = (unsigned int)(pres >> 32);
        lo |= __shfl_xor(lo, off, 32);
        hi |= __shfl_xor(hi, off, 32);
        pres = ((unsigned long long)hi << 32) | (unsigned long long)lo;
    }
    if (lane == 0)
        uniq[wid] = (float)__builtin_popcountll(pres) * (1.0f / (float)OBS_);
}

// ---------------------------------------------------------------------------
// Phase 3: feature assembly + both MLPs with v_wmma_f32_16x16x32_f16.
// One block handles 16 batch rows; 8 waves = 8 N-tiles of HID=128.
// ---------------------------------------------------------------------------
__global__ void phase3_kernel(const float* __restrict__ psig, const float* __restrict__ scal,
                              const float* __restrict__ uniq,
                              const float* __restrict__ zw1, const float* __restrict__ zb1,
                              const float* __restrict__ zw2, const float* __restrict__ zb2,
                              const float* __restrict__ yw1, const float* __restrict__ yb1,
                              const float* __restrict__ yw2, const float* __restrict__ yb2,
                              float* __restrict__ out)
{
    __shared__ float featA[16][64];   // feat padded 52 -> 64
    __shared__ float hBuf[16][HID_];
    __shared__ float zBuf[16][16];    // cols 0..7 logits, cols 8..15 straight-through z
    __shared__ float feat2[16][32];   // feat2 padded 24 -> 32

    int tid  = threadIdx.x;
    int lane = tid & 31;
    int wave = tid >> 5;
    int brow = blockIdx.x * 16;

    // feat = [p_sig(16) | uniq_norm(32) | ent | top1 | top2 | mass | 0-pad]
    for (int idx = tid; idx < 16 * 64; idx += blockDim.x) {
        int r = idx >> 6, c = idx & 63;
        int b = brow + r;
        float v;
        if      (c < 16) v = psig[b * Y_ + c];
        else if (c < 48) v = uniq[b * V_ + (c - 16)];
        else if (c < 52) v = scal[b * 4 + (c - 48)];
        else             v = 0.f;
        featA[r][c] = v;
    }
    __syncthreads();

    // GEMM1: feat(16x64) @ zw1(52x128), K padded to 64 (2 WMMA), gelu -> hBuf
    {
        int nt = wave;
        v16h a0 = load_a_frag(&featA[0][0], 64, 0,  lane);
        v16h a1 = load_a_frag(&featA[0][0], 64, 32, lane);
        v16h w0 = load_b_frag(zw1, HID_, 52, HID_, 0,  nt * 16, lane);
        v16h w1 = load_b_frag(zw1, HID_, 52, HID_, 32, nt * 16, lane);
        v8f c = {};
        c = __builtin_amdgcn_wmma_f32_16x16x32_f16(false, a0, false, w0, (short)0, c, false, false);
        c = __builtin_amdgcn_wmma_f32_16x16x32_f16(false, a1, false, w1, (short)0, c, false, false);
        int n  = nt * 16 + (lane & 15);
        int mb = (lane >> 4) * 8;
#pragma unroll
        for (int r = 0; r < 8; ++r)
            hBuf[mb + r][n] = gelu_exact(c[r] + zb1[n]);
    }
    __syncthreads();

    // GEMM2: h(16x128) @ zw2(128x8), N padded to 16 (wave 0 only)
    if (wave == 0) {
        v8f c = {};
#pragma unroll
        for (int kc = 0; kc < HID_; kc += 32) {
            v16h a = load_a_frag(&hBuf[0][0], HID_, kc, lane);
            v16h w = load_b_frag(zw2, 8, HID_, 8, kc, 0, lane);
            c = __builtin_amdgcn_wmma_f32_16x16x32_f16(false, a, false, w, (short)0, c, false, false);
        }
        int n  = lane & 15;
        int mb = (lane >> 4) * 8;
        if (n < 8) {
#pragma unroll
            for (int r = 0; r < 8; ++r) zBuf[mb + r][n] = c[r] + zb2[n];
        }
    }
    __syncthreads();

    // straight-through z per row: forward value = one_hot(argmax) - softmax + softmax
    if (tid < 16) {
        float lmax = zBuf[tid][0]; int am = 0;
        for (int j = 1; j < 8; ++j) { float v = zBuf[tid][j]; if (v > lmax) { lmax = v; am = j; } }
        float e[8], s = 0.f;
        for (int j = 0; j < 8; ++j) { e[j] = expf(zBuf[tid][j] - lmax); s += e[j]; }
        float invs = 1.f / s;
        for (int j = 0; j < 8; ++j) {
            float soft = e[j] * invs;
            float hard = (j == am) ? 1.f : 0.f;
            zBuf[tid][8 + j] = (hard - soft) + soft;
        }
    }
    __syncthreads();

    // feat2 = [p_sig(16) | z(8) | 0-pad(8)]
    for (int idx = tid; idx < 16 * 32; idx += blockDim.x) {
        int r = idx >> 5, c = idx & 31;
        float v;
        if      (c < 16) v = psig[(brow + r) * Y_ + c];
        else if (c < 24) v = zBuf[r][8 + (c - 16)];
        else             v = 0.f;
        feat2[r][c] = v;
    }
    __syncthreads();

    // GEMM3: feat2(16x32) @ yw1(24x128), gelu -> hBuf
    {
        int nt = wave;
        v16h a = load_a_frag(&feat2[0][0], 32, 0, lane);
        v16h w = load_b_frag(yw1, HID_, 24, HID_, 0, nt * 16, lane);
        v8f c = {};
        c = __builtin_amdgcn_wmma_f32_16x16x32_f16(false, a, false, w, (short)0, c, false, false);
        int n  = nt * 16 + (lane & 15);
        int mb = (lane >> 4) * 8;
#pragma unroll
        for (int r = 0; r < 8; ++r)
            hBuf[mb + r][n] = gelu_exact(c[r] + yb1[n]);
    }
    __syncthreads();

    // GEMM4: h(16x128) @ yw2(128x16) -> out (wave 0 only)
    if (wave == 0) {
        v8f c = {};
#pragma unroll
        for (int kc = 0; kc < HID_; kc += 32) {
            v16h a = load_a_frag(&hBuf[0][0], HID_, kc, lane);
            v16h w = load_b_frag(yw2, 16, HID_, 16, kc, 0, lane);
            c = __builtin_amdgcn_wmma_f32_16x16x32_f16(false, a, false, w, (short)0, c, false, false);
        }
        int n  = lane & 15;
        int mb = (lane >> 4) * 8;
#pragma unroll
        for (int r = 0; r < 8; ++r)
            out[(brow + mb + r) * Y_ + n] = c[r] + yb2[n];
    }
}

// ---------------------------------------------------------------------------
extern "C" void kernel_launch(void* const* d_in, const int* in_sizes, int n_in,
                              void* d_out, int out_size, void* d_ws, size_t ws_size,
                              hipStream_t stream) {
    (void)n_in; (void)out_size;
    const int*   tables    = (const int*)d_in[0];
    const int*   sigma     = (const int*)d_in[1];
    const int*   base_obs  = (const int*)d_in[2];
    const int*   actions   = (const int*)d_in[3];
    const int*   responses = (const int*)d_in[4];
    // d_in[5] is t; the reference treats it as a static small step count = T below
    const float* zw1 = (const float*)d_in[6];
    const float* zb1 = (const float*)d_in[7];
    const float* zw2 = (const float*)d_in[8];
    const float* zb2 = (const float*)d_in[9];
    const float* yw1 = (const float*)d_in[10];
    const float* yb1 = (const float*)d_in[11];
    const float* yw2 = (const float*)d_in[12];
    const float* yb2 = (const float*)d_in[13];
    float* out = (float*)d_out;

    const int B = in_sizes[2];                  // base_obs is (B,)
    const int T = (B > 0) ? in_sizes[3] / B : 0; // actions is (B,T)

    char* ws = (char*)d_ws;
    size_t off = 0;
    unsigned int*   candWs = (unsigned int*)(ws + off);   off += (size_t)B * NW_ * 4;
    float*          psig   = (float*)(ws + off);          off += (size_t)B * Y_ * 4;
    float*          scal   = (float*)(ws + off);          off += (size_t)B * 4 * 4;
    float*          uniq   = (float*)(ws + off);          off += (size_t)B * V_ * 4;
    int*            cntWs  = (int*)(ws + off);            off += (size_t)B * 4;
    unsigned short* listWs = (unsigned short*)(ws + off);
    size_t need = off + (size_t)B * N_ * 2;
    int haveList = (ws_size >= need) ? 1 : 0;   // deterministic for a fixed run

    phase1_kernel<<<B, 256, 0, stream>>>(tables, sigma, base_obs, actions, responses, T,
                                         candWs, psig, scal, cntWs, listWs, haveList);

    const int pairs = B * V_;
    phase2_kernel<<<(pairs + 7) / 8, 256, 0, stream>>>(tables, candWs, cntWs, listWs,
                                                       uniq, haveList, pairs);

    phase3_kernel<<<B / 16, 256, 0, stream>>>(psig, scal, uniq,
                                              zw1, zb1, zw2, zb2,
                                              yw1, yb1, yw2, yb2, out);
}
